// CoordsToToken_67345087201368
// MI455X (gfx1250) — compile-verified, matrix operand order checked by reading, not run on hardware
//
#include <hip/hip_runtime.h>

typedef __attribute__((ext_vector_type(16))) _Float16 v16h;
typedef __attribute__((ext_vector_type(8)))  float    v8f;

#define N_TOK 20

// ds_swizzle group-of-32: offset = {1'b0, xor[4:0], or[4:0], and[4:0]}
#define SWIZ(x, imm)   __builtin_amdgcn_ds_swizzle((x), (imm))
#define SWIZ_XOR(x, m) __builtin_amdgcn_ds_swizzle((x), (((m) << 10) | 0x1f))

__device__ __forceinline__ int imin2(int a, int b) { return a < b ? a : b; }

// One 16-residue group: build A from packed counts, run the two token tiles,
// then a folding multi-row min-butterfly (15 swizzles for all 16 rows).
// Each stage issues all its swizzles into distinct temps before consuming,
// so the DS ops overlap and each stage pays one wait, not one per element.
// Key = 32*score + token: lower score wins, ties break to the lower token.
template<int GSWZ>
__device__ __forceinline__ void classify_group(
    int p01, int p23, bool hi, int lane,
    const v16h& B1, const v16h& B2, const v8f& C1, const v8f& C2,
    float* __restrict__ out, int resBase)
{
  // A (16x32 f16): lanes 0-15 hold row M=lane, K=0..3 = counts; rest zero.
  int q01 = SWIZ(p01, GSWZ);
  int q23 = SWIZ(p23, GSWZ);
  q01 = hi ? 0 : q01;
  q23 = hi ? 0 : q23;
  union { int i[8]; v16h h; } au;
#pragma unroll
  for (int i = 0; i < 8; ++i) au.i[i] = 0;
  au.i[0] = q01; au.i[1] = q23;
  const v16h A = au.h;

  // D[r][t] = 32*(||m_t||^2 - 2*c_r.m_t) + t  (exact small integers in f32)
  v8f D1 = __builtin_amdgcn_wmma_f32_16x16x32_f16(false, A, false, B1,
                                                  (short)0, C1, false, false);
  v8f D2 = __builtin_amdgcn_wmma_f32_16x16x32_f16(false, A, false, B2,
                                                  (short)0, C2, false, false);

  int k[8];
#pragma unroll
  for (int v = 0; v < 8; ++v) k[v] = imin2((int)D1[v], (int)D2[v]);

  // Stage 1: reduce cols over {c, c^8}; 8 swizzles in flight, then mins.
  int w[8];
#pragma unroll
  for (int v = 0; v < 8; ++v) w[v] = SWIZ_XOR(k[v], 8);
#pragma unroll
  for (int v = 0; v < 8; ++v) k[v] = imin2(k[v], w[v]);
  const bool b3 = (lane & 8) != 0;           // fold: rows v / v+4
  int t0 = b3 ? k[4] : k[0];
  int t1 = b3 ? k[5] : k[1];
  int t2 = b3 ? k[6] : k[2];
  int t3 = b3 ? k[7] : k[3];

  // Stage 2: reduce over {c, c^4} (c^4 preserves the fold bit 3).
  int u0 = SWIZ_XOR(t0, 4);
  int u1 = SWIZ_XOR(t1, 4);
  int u2 = SWIZ_XOR(t2, 4);
  int u3 = SWIZ_XOR(t3, 4);
  t0 = imin2(t0, u0);
  t1 = imin2(t1, u1);
  t2 = imin2(t2, u2);
  t3 = imin2(t3, u3);
  const bool b2 = (lane & 4) != 0;           // fold: rows +2
  int s0 = b2 ? t2 : t0;
  int s1 = b2 ? t3 : t1;

  // Stage 3: reduce over {c, c^2}.
  int v0 = SWIZ_XOR(s0, 2);
  int v1 = SWIZ_XOR(s1, 2);
  s0 = imin2(s0, v0);
  s1 = imin2(s1, v1);
  const bool b1 = (lane & 2) != 0;           // fold: rows +1
  int r = b1 ? s1 : s0;

  // Stage 4: reduce over {c, c^1} -> full 16-column min.
  r = imin2(r, SWIZ_XOR(r, 1));

  // Lane l now holds row ((l>>1)&7) + 8*hi; even lanes emit all 16 rows.
  if ((lane & 1) == 0) {
    const int row = ((lane >> 1) & 7) + (hi ? 8 : 0);
    out[resBase + row] = (float)(r & 31);
  }
}

__global__ __launch_bounds__(256) void coords_to_token_kernel(
    const float* __restrict__ coords,   // [N_ATOMS][3] f32
    const int*   __restrict__ des_idx,  // [N_RES*14]
    const float* __restrict__ cm,       // [20][4] f32 (small integers 0..9)
    float*       __restrict__ out)      // [N_RES] token id as f32
{
  const int  tid      = blockIdx.x * blockDim.x + threadIdx.x;
  const int  lane     = threadIdx.x & 31;
  const int  col      = lane & 15;
  const bool hi       = lane >= 16;
  const int  waveBase = tid & ~31;

  // ---------------- scalar gather + contact counting ----------------
  const long long base = (long long)tid * 14;
  float bx[4], by[4], bz[4];
#pragma unroll
  for (int i = 0; i < 4; ++i) {
    const float* p = coords + 3ll * (long long)des_idx[base + i];
    bx[i] = p[0]; by[i] = p[1]; bz[i] = p[2];
  }
  float c0 = 0.f, c1 = 0.f, c2 = 0.f, c3 = 0.f;
#pragma unroll
  for (int j = 4; j < 14; ++j) {
    const float* p = coords + 3ll * (long long)des_idx[base + j];
    const float sx = p[0], sy = p[1], sz = p[2];
    float d0, d1, d2, d3;
    { float dx = bx[0]-sx, dy = by[0]-sy, dz = bz[0]-sz; d0 = fmaf(dx,dx,fmaf(dy,dy,dz*dz)); }
    { float dx = bx[1]-sx, dy = by[1]-sy, dz = bz[1]-sz; d1 = fmaf(dx,dx,fmaf(dy,dy,dz*dz)); }
    { float dx = bx[2]-sx, dy = by[2]-sy, dz = bz[2]-sz; d2 = fmaf(dx,dx,fmaf(dy,dy,dz*dz)); }
    { float dx = bx[3]-sx, dy = by[3]-sy, dz = bz[3]-sz; d3 = fmaf(dx,dx,fmaf(dy,dy,dz*dz)); }
    const float m  = fminf(fminf(d0, d1), fminf(d2, d3));
    const bool  ok = (m <= 0.25f);             // sqrt(d2) <= THRESHOLD=0.5
    const bool  i0 = (d0 == m);                // first-occurrence argmin
    const bool  i1 = !i0 && (d1 == m);
    const bool  i2 = !i0 && !i1 && (d2 == m);
    const bool  i3 = !(i0 | i1 | i2);
    c0 += (ok & i0) ? 1.f : 0.f;
    c1 += (ok & i1) ? 1.f : 0.f;
    c2 += (ok & i2) ? 1.f : 0.f;
    c3 += (ok & i3) ? 1.f : 0.f;
  }

  // Pack the 4 counts into two f16x2 words (exact: integers <= 10).
  const int p01 = __builtin_bit_cast(int, __builtin_amdgcn_cvt_pkrtz(c0, c1));
  const int p23 = __builtin_bit_cast(int, __builtin_amdgcn_cvt_pkrtz(c2, c3));

  // ---- loop-invariant operands: B = -64*M^T, C = 32*||m_t||^2 + t ----
  // B (32x16 f16): lanes 0-15 carry K=0..3 of column N=lane; K>=4 zero;
  // lanes 16-31 (K=8..15, 24..31) all zero.
  v16h B1 = {}; v16h B2 = {};
  float bias1, bias2;
  {
    const int t1 = col;                        // tokens 0..15
    const float m0 = cm[t1*4+0], m1 = cm[t1*4+1], m2 = cm[t1*4+2], m3 = cm[t1*4+3];
    bias1 = 32.f*(m0*m0 + m1*m1 + m2*m2 + m3*m3) + (float)t1;
    if (!hi) {
      B1[0] = (_Float16)(-64.f*m0); B1[1] = (_Float16)(-64.f*m1);
      B1[2] = (_Float16)(-64.f*m2); B1[3] = (_Float16)(-64.f*m3);
    }
    const int t2 = 16 + col;                   // tokens 16..19, rest padded
    if (t2 < N_TOK) {
      const float n0 = cm[t2*4+0], n1 = cm[t2*4+1], n2 = cm[t2*4+2], n3 = cm[t2*4+3];
      bias2 = 32.f*(n0*n0 + n1*n1 + n2*n2 + n3*n3) + (float)t2;
      if (!hi) {
        B2[0] = (_Float16)(-64.f*n0); B2[1] = (_Float16)(-64.f*n1);
        B2[2] = (_Float16)(-64.f*n2); B2[3] = (_Float16)(-64.f*n3);
      }
    } else {
      bias2 = 1.0e6f;                          // padding columns can never win
    }
  }
  v8f C1, C2;
#pragma unroll
  for (int v = 0; v < 8; ++v) { C1[v] = bias1; C2[v] = bias2; }

  // ---- matrix phase: residues of lanes 0-15 (src = l&15), then 16-31 ----
  classify_group<0x00f>(p01, p23, hi, lane, B1, B2, C1, C2, out, waveBase);
  classify_group<0x20f>(p01, p23, hi, lane, B1, B2, C1, C2, out, waveBase + 16);
}

extern "C" void kernel_launch(void* const* d_in, const int* in_sizes, int n_in,
                              void* d_out, int out_size, void* d_ws, size_t ws_size,
                              hipStream_t stream) {
  (void)in_sizes; (void)n_in; (void)d_ws; (void)ws_size;
  const float* coords = (const float*)d_in[0];
  const int*   des    = (const int*)d_in[1];   // harness stages integers as int32
  const float* cm     = (const float*)d_in[2];
  float*       out    = (float*)d_out;

  const int nres  = out_size;                  // 262144, divisible by 256
  const int block = 256;                       // 8 wave32s per block
  const int grid  = nres / block;              // exact cover -> EXEC all-ones for WMMA
  coords_to_token_kernel<<<grid, block, 0, stream>>>(coords, des, cm, out);
}
